// GraphNN_58471684768101
// MI455X (gfx1250) — matrix operand order repaired and checked by reading, not compiled, not run on hardware
//
#include <hip/hip_runtime.h>
#include <math.h>

// Problem constants (from reference): J=4096, C=8192, F=64, K=2
constexpr int JN = 4096;
constexpr int CN = 8192;
constexpr int FN = 64;
#define TOTAL ((size_t)JN * (size_t)CN)

constexpr int COL_SLICES  = 16;   // col-sum reduction split (256 rows each)
constexpr int PROP_SLICES = 32;   // prop reduction split (128 rows each)

typedef _Float16 h16v __attribute__((ext_vector_type(16)));
typedef _Float16 h8v  __attribute__((ext_vector_type(8)));
typedef float    f8v  __attribute__((ext_vector_type(8)));

// ---------------------------------------------------------------------------
// 1) Stable bitonic sort of (h,L) by (h desc, idx asc). Single WGP, LDS.
// ---------------------------------------------------------------------------
__global__ __launch_bounds__(1024)
void sort_kernel(const float* __restrict__ h, const float* __restrict__ L,
                 float* __restrict__ hs, float* __restrict__ Ls) {
  __shared__ float sh[JN];
  __shared__ float sl[JN];
  __shared__ int   si[JN];
  const int tid = threadIdx.x;
  for (int i = tid; i < JN; i += 1024) { sh[i] = h[i]; sl[i] = L[i]; si[i] = i; }
  __syncthreads();
  for (int k = 2; k <= JN; k <<= 1) {
    for (int j = k >> 1; j > 0; j >>= 1) {
      for (int i = tid; i < JN; i += 1024) {
        int l = i ^ j;
        if (l > i) {
          float ha = sh[i], hb = sh[l];
          int   ia = si[i], ib = si[l];
          // "after" in target order (h desc, idx asc)
          bool after = (ha < hb) || (ha == hb && ia > ib);
          bool sw = ((i & k) == 0) ? after : !after;
          if (sw) {
            sh[i] = hb; sh[l] = ha;
            float t = sl[i]; sl[i] = sl[l]; sl[l] = t;
            si[i] = ib; si[l] = ia;
          }
        }
      }
      __syncthreads();
    }
  }
  for (int i = tid; i < JN; i += 1024) { hs[i] = sh[i]; Ls[i] = sl[i]; }
}

// ---------------------------------------------------------------------------
// 2) deg/norm: column sums of Graph, split over COL_SLICES row-slices so the
//    128 MB stream has ~131K threads in flight (deterministic fixed-order).
// ---------------------------------------------------------------------------
__global__ __launch_bounds__(256)
void col_sum_part_kernel(const float* __restrict__ Graph, float* __restrict__ colpart) {
  int c = blockIdx.x * 256 + threadIdx.x;
  int s = blockIdx.y;
  int j0 = s * (JN / COL_SLICES);
  float acc = 0.f;
  for (int j = j0; j < j0 + (JN / COL_SLICES); ++j) acc += Graph[(size_t)j * CN + c];
  colpart[(size_t)s * CN + c] = acc;
}

__global__ __launch_bounds__(256)
void col_sum_final_kernel(const float* __restrict__ colpart, float* __restrict__ degv,
                          float* __restrict__ nrm) {
  int c = blockIdx.x * 256 + threadIdx.x;
  float acc = 0.f;
#pragma unroll
  for (int s = 0; s < COL_SLICES; ++s) acc += colpart[(size_t)s * CN + c];
  degv[c] = acc;                        // exact: sums of 0/1 ints
  nrm[c]  = rsqrtf(fmaxf(acc, 1.0f));   // clip(deg,1)^-0.5
}

// ---------------------------------------------------------------------------
// 3) row sums of Graph (one block per row)
// ---------------------------------------------------------------------------
__global__ __launch_bounds__(256)
void row_sum_kernel(const float* __restrict__ Graph, float* __restrict__ rowv) {
  __shared__ float s[256];
  int j = blockIdx.x;
  float acc = 0.f;
  for (int c = threadIdx.x; c < CN; c += 256) acc += Graph[(size_t)j * CN + c];
  s[threadIdx.x] = acc; __syncthreads();
  for (int off = 128; off > 0; off >>= 1) {
    if (threadIdx.x < off) s[threadIdx.x] += s[threadIdx.x + off];
    __syncthreads();
  }
  if (threadIdx.x == 0) rowv[j] = s[0];
}

// ---------------------------------------------------------------------------
// 4) Y = norm[j] * X[j]   (X = [hs, Ls, W, P, N, 1])
// ---------------------------------------------------------------------------
__global__ __launch_bounds__(256)
void build_y_kernel(const float* __restrict__ hs, const float* __restrict__ Ls,
                    const float* __restrict__ W, const float* __restrict__ P,
                    const float* __restrict__ N, const float* __restrict__ nrm,
                    float* __restrict__ Y) {
  int j = blockIdx.x * 256 + threadIdx.x;
  if (j >= JN) return;
  float nj = nrm[j];
  Y[j * 6 + 0] = nj * hs[j];
  Y[j * 6 + 1] = nj * Ls[j];
  Y[j * 6 + 2] = nj * W[0];
  Y[j * 6 + 3] = nj * P[0];
  Y[j * 6 + 4] = nj * N[0];
  Y[j * 6 + 5] = nj;
}

__global__ __launch_bounds__(256)
void scale_y_kernel(const float* __restrict__ Fin, const float* __restrict__ nrm,
                    float* __restrict__ Y) {
  int i = blockIdx.x * 256 + threadIdx.x;
  if (i >= JN * 6) return;
  Y[i] = nrm[i / 6] * Fin[i];
}

// ---------------------------------------------------------------------------
// 5) prop: Fout[c,:] = norm[c] * sum_j Graph[j,c] * Y[j,:]  for c < JN.
//    Reduction over j split into PROP_SLICES deterministic partials.
// ---------------------------------------------------------------------------
__global__ __launch_bounds__(256)
void prop_part_kernel(const float* __restrict__ Graph, const float* __restrict__ Y,
                      float* __restrict__ proppart) {
  int c = blockIdx.x * 256 + threadIdx.x;   // c < JN
  int s = blockIdx.y;
  int j0 = s * (JN / PROP_SLICES);
  float a0 = 0.f, a1 = 0.f, a2 = 0.f, a3 = 0.f, a4 = 0.f, a5 = 0.f;
  for (int j = j0; j < j0 + (JN / PROP_SLICES); ++j) {
    float g = Graph[(size_t)j * CN + c];            // coalesced across lanes
    const float* y = Y + j * 6;                     // wave-uniform -> scalar loads
    a0 = fmaf(g, y[0], a0); a1 = fmaf(g, y[1], a1); a2 = fmaf(g, y[2], a2);
    a3 = fmaf(g, y[3], a3); a4 = fmaf(g, y[4], a4); a5 = fmaf(g, y[5], a5);
  }
  float* p = proppart + ((size_t)s * JN + c) * 6;
  p[0] = a0; p[1] = a1; p[2] = a2; p[3] = a3; p[4] = a4; p[5] = a5;
}

__global__ __launch_bounds__(256)
void prop_final_kernel(const float* __restrict__ proppart, const float* __restrict__ nrm,
                       float* __restrict__ Fout) {
  int c = blockIdx.x * 256 + threadIdx.x;
  if (c >= JN) return;
  float a[6] = {0.f, 0.f, 0.f, 0.f, 0.f, 0.f};
  for (int s = 0; s < PROP_SLICES; ++s) {          // fixed order: deterministic
    const float* p = proppart + ((size_t)s * JN + c) * 6;
#pragma unroll
    for (int t = 0; t < 6; ++t) a[t] += p[t];
  }
  float nc = nrm[c];
#pragma unroll
  for (int t = 0; t < 6; ++t) Fout[c * 6 + t] = nc * a[t];
}

// ---------------------------------------------------------------------------
// 6) G_t[:J] = [X|f1|f2] @ tag_w + tag_b, emit as f16 A-matrix (J x 64)
// ---------------------------------------------------------------------------
__global__ __launch_bounds__(256)
void gt_kernel(const float* __restrict__ hs, const float* __restrict__ Ls,
               const float* __restrict__ W, const float* __restrict__ P,
               const float* __restrict__ N, const float* __restrict__ f1,
               const float* __restrict__ f2, const float* __restrict__ tag_w,
               const float* __restrict__ tag_b, _Float16* __restrict__ A16) {
  int id = blockIdx.x * 256 + threadIdx.x;       // JN*FN threads
  if (id >= JN * FN) return;
  int j = id >> 6, f = id & 63;
  float feat[18];
  feat[0] = hs[j]; feat[1] = Ls[j]; feat[2] = W[0]; feat[3] = P[0];
  feat[4] = N[0];  feat[5] = 1.0f;
#pragma unroll
  for (int t = 0; t < 6; ++t) { feat[6 + t] = f1[j * 6 + t]; feat[12 + t] = f2[j * 6 + t]; }
  float acc = tag_b[f];
#pragma unroll
  for (int t = 0; t < 18; ++t) acc = fmaf(feat[t], tag_w[t * FN + f], acc);
  A16[id] = (_Float16)acc;
}

// ---------------------------------------------------------------------------
// 7) Bt[c,k] = (f16) lin_w[k,c]   -> contiguous-K B operand for WMMA
// ---------------------------------------------------------------------------
__global__ __launch_bounds__(256)
void bt_kernel(const float* __restrict__ lin_w, _Float16* __restrict__ Bt) {
  int id = blockIdx.x * 256 + threadIdx.x;       // FN*CN threads
  if (id >= FN * CN) return;
  int k = id >> 13, c = id & (CN - 1);
  Bt[(size_t)c * FN + k] = (_Float16)lin_w[id];  // coalesced read
}

// ---------------------------------------------------------------------------
// 8) Value = A16 (4096x64) * Bt^T (64x8192) + lin_b, via v_wmma_f32_16x16x32_f16
//    block = 8 waves sharing one 16-col strip; B tile staged once in LDS.
// ---------------------------------------------------------------------------
__global__ __launch_bounds__(256)
void value_gemm_kernel(const _Float16* __restrict__ A,   // (JN, 64)
                       const _Float16* __restrict__ Bt,  // (CN, 64)
                       const float* __restrict__ lin_b,  // (CN)
                       float* __restrict__ Value) {      // (JN, CN)
  __shared__ __align__(16) _Float16 Bs[16 * FN];         // 2 KB B tile
  const int wave = threadIdx.x >> 5;
  const int lane = threadIdx.x & 31;
  const int n    = lane & 15;
  const int hw   = lane >> 4;                 // half-wave 0/1
  const int ct   = blockIdx.x;                // 0..CN/16-1
  const int rt   = blockIdx.y * 8 + wave;     // 0..JN/16-1
  const int col  = ct * 16 + n;
  const int m    = rt * 16 + n;

  // Stage the shared B tile (cols ct*16..+15 are contiguous rows of Bt).
  {
    const float* src = (const float*)(Bt + (size_t)ct * 16 * FN);  // 512 floats
    float* dst = (float*)Bs;
    dst[threadIdx.x]       = src[threadIdx.x];
    dst[threadIdx.x + 256] = src[threadIdx.x + 256];
  }
  __syncthreads();

  // A: lane holds row m; K = {hw*8..+7, 16+hw*8..+7} (+32/+48 for 2nd WMMA)
  const _Float16* Ar = A + (size_t)m * FN;
  h8v a0lo = *(const h8v*)(Ar +      hw * 8);
  h8v a0hi = *(const h8v*)(Ar + 16 + hw * 8);
  h8v a1lo = *(const h8v*)(Ar + 32 + hw * 8);
  h8v a1hi = *(const h8v*)(Ar + 48 + hw * 8);
  // B (from LDS): lane holds col; K = {hw*8..+7, 16+hw*8..+7} (+32 for 2nd WMMA)
  const _Float16* Br = Bs + n * FN + hw * 8;
  h8v b0lo = *(const h8v*)(Br + 0);
  h8v b0hi = *(const h8v*)(Br + 16);
  h8v b1lo = *(const h8v*)(Br + 32);
  h8v b1hi = *(const h8v*)(Br + 48);

  h16v a0, a1, b0, b1;
#pragma unroll
  for (int e = 0; e < 8; ++e) {
    a0[e] = a0lo[e]; a0[8 + e] = a0hi[e];
    a1[e] = a1lo[e]; a1[8 + e] = a1hi[e];
    b0[e] = b0lo[e]; b0[8 + e] = b0hi[e];
    b1[e] = b1lo[e]; b1[8 + e] = b1hi[e];
  }
  f8v c = {};
  c = __builtin_amdgcn_wmma_f32_16x16x32_f16(false, a0, false, b0, (short)0, c, false, false);
  c = __builtin_amdgcn_wmma_f32_16x16x32_f16(false, a1, false, b1, (short)0, c, false, false);

  float bias = lin_b[col];
#pragma unroll
  for (int r = 0; r < 8; ++r) {
    int rowi = rt * 16 + hw * 8 + r;          // D layout: VGPR r -> row hw*8+r
    Value[(size_t)rowi * CN + col] = c[r] + bias;
  }
}

// ---------------------------------------------------------------------------
// 9) masked logits + global softmax (3 streaming passes; mask recomputed)
// ---------------------------------------------------------------------------
__device__ __forceinline__
float compute_tmp(size_t idx, const float* __restrict__ Value,
                  const float* __restrict__ rowv, const float* __restrict__ degv) {
  int j = (int)(idx >> 13);
  int c = (int)(idx & (CN - 1));
  float V  = Value[idx];
  float rj = rowv[j];
  float maskv;
  if (c < JN) {
    float left = 1.0f - rj - rowv[c] - degv[c] - degv[j];  // exact int arith
    maskv = (left == 1.0f && c > j) ? 1.0f : 0.0f;         // triu(k=1)
  } else {
    maskv = 1.0f - rj;
  }
  return V - (1.0f - maskv) * 10000.0f;
}

__global__ __launch_bounds__(256)
void tmp_max_kernel(const float* __restrict__ Value, const float* __restrict__ rowv,
                    const float* __restrict__ degv, float* __restrict__ partial) {
  __shared__ float s[256];
  float m = -3.402823466e38f;
  for (size_t idx = (size_t)blockIdx.x * 256 + threadIdx.x; idx < TOTAL;
       idx += (size_t)gridDim.x * 256)
    m = fmaxf(m, compute_tmp(idx, Value, rowv, degv));
  s[threadIdx.x] = m; __syncthreads();
  for (int off = 128; off > 0; off >>= 1) {
    if (threadIdx.x < off) s[threadIdx.x] = fmaxf(s[threadIdx.x], s[threadIdx.x + off]);
    __syncthreads();
  }
  if (threadIdx.x == 0) partial[blockIdx.x] = s[0];
}

__global__ __launch_bounds__(1024)
void final_max_kernel(const float* __restrict__ partial, float* __restrict__ out) {
  __shared__ float s[1024];
  s[threadIdx.x] = fmaxf(partial[threadIdx.x], partial[threadIdx.x + 1024]);
  __syncthreads();
  for (int off = 512; off > 0; off >>= 1) {
    if (threadIdx.x < off) s[threadIdx.x] = fmaxf(s[threadIdx.x], s[threadIdx.x + off]);
    __syncthreads();
  }
  if (threadIdx.x == 0) out[0] = s[0];
}

__global__ __launch_bounds__(256)
void tmp_expsum_kernel(const float* __restrict__ Value, const float* __restrict__ rowv,
                       const float* __restrict__ degv, const float* __restrict__ maxv,
                       float* __restrict__ partial) {
  __shared__ float s[256];
  float mv = maxv[0];
  float acc = 0.f;
  for (size_t idx = (size_t)blockIdx.x * 256 + threadIdx.x; idx < TOTAL;
       idx += (size_t)gridDim.x * 256)
    acc += expf(compute_tmp(idx, Value, rowv, degv) - mv);
  s[threadIdx.x] = acc; __syncthreads();
  for (int off = 128; off > 0; off >>= 1) {
    if (threadIdx.x < off) s[threadIdx.x] += s[threadIdx.x + off];
    __syncthreads();
  }
  if (threadIdx.x == 0) partial[blockIdx.x] = s[0];
}

__global__ __launch_bounds__(1024)
void final_sum_kernel(const float* __restrict__ partial, float* __restrict__ out) {
  __shared__ float s[1024];
  s[threadIdx.x] = partial[threadIdx.x] + partial[threadIdx.x + 1024];
  __syncthreads();
  for (int off = 512; off > 0; off >>= 1) {
    if (threadIdx.x < off) s[threadIdx.x] += s[threadIdx.x + off];
    __syncthreads();
  }
  if (threadIdx.x == 0) out[0] = s[0];
}

__global__ __launch_bounds__(256)
void poss_kernel(const float* __restrict__ Value, const float* __restrict__ rowv,
                 const float* __restrict__ degv, const float* __restrict__ maxv,
                 const float* __restrict__ sumv, float* __restrict__ Poss) {
  float mv  = maxv[0];
  float inv = 1.0f / sumv[0];
  for (size_t idx = (size_t)blockIdx.x * 256 + threadIdx.x; idx < TOTAL;
       idx += (size_t)gridDim.x * 256)
    Poss[idx] = expf(compute_tmp(idx, Value, rowv, degv) - mv) * inv;
}

// ---------------------------------------------------------------------------
extern "C" void kernel_launch(void* const* d_in, const int* in_sizes, int n_in,
                              void* d_out, int out_size, void* d_ws, size_t ws_size,
                              hipStream_t stream) {
  (void)in_sizes; (void)n_in; (void)out_size; (void)ws_size;
  const float* h     = (const float*)d_in[0];
  const float* L     = (const float*)d_in[1];
  const float* W     = (const float*)d_in[2];
  const float* P     = (const float*)d_in[3];
  const float* N     = (const float*)d_in[4];
  const float* Graph = (const float*)d_in[5];
  const float* tag_w = (const float*)d_in[6];
  const float* tag_b = (const float*)d_in[7];
  const float* lin_w = (const float*)d_in[8];
  const float* lin_b = (const float*)d_in[9];

  float* out   = (float*)d_out;
  float* Value = out;                 // first output
  float* Poss  = out + TOTAL;         // second output

  // workspace layout (float units)
  float* ws   = (float*)d_ws;
  float* hs   = ws + 0;        // 4096
  float* Ls   = ws + 4096;     // 4096
  float* degv = ws + 8192;     // 8192
  float* nrm  = ws + 16384;    // 8192
  float* rowv = ws + 24576;    // 4096
  float* Y    = ws + 28672;    // 24576
  float* f1   = ws + 53248;    // 24576
  float* f2   = ws + 77824;    // 24576
  float* red  = ws + 102400;   // 2048 partials + 2 finals (finals overlap A16 start at 104448)
  float* maxv = red + 2048;
  float* sumv = red + 2049;
  _Float16* A16 = (_Float16*)(ws + 104448);   // 4096*64 halfs (131072 float slots)
  _Float16* Bt  = (_Float16*)(ws + 235520);   // 8192*64 halfs (262144 float slots)
  float* colpart  = ws + 497664;              // 16*8192   = 131072
  float* proppart = ws + 628736;              // 32*4096*6 = 786432 (end 1415168)

  sort_kernel<<<1, 1024, 0, stream>>>(h, L, hs, Ls);
  col_sum_part_kernel<<<dim3(CN / 256, COL_SLICES), 256, 0, stream>>>(Graph, colpart);
  col_sum_final_kernel<<<CN / 256, 256, 0, stream>>>(colpart, degv, nrm);
  row_sum_kernel<<<JN, 256, 0, stream>>>(Graph, rowv);

  build_y_kernel<<<JN / 256, 256, 0, stream>>>(hs, Ls, W, P, N, nrm, Y);
  prop_part_kernel<<<dim3(JN / 256, PROP_SLICES), 256, 0, stream>>>(Graph, Y, proppart);
  prop_final_kernel<<<JN / 256, 256, 0, stream>>>(proppart, nrm, f1);
  scale_y_kernel<<<(JN * 6 + 255) / 256, 256, 0, stream>>>(f1, nrm, Y);
  prop_part_kernel<<<dim3(JN / 256, PROP_SLICES), 256, 0, stream>>>(Graph, Y, proppart);
  prop_final_kernel<<<JN / 256, 256, 0, stream>>>(proppart, nrm, f2);

  gt_kernel<<<(JN * FN) / 256, 256, 0, stream>>>(hs, Ls, W, P, N, f1, f2, tag_w, tag_b, A16);
  bt_kernel<<<(FN * CN) / 256, 256, 0, stream>>>(lin_w, Bt);

  value_gemm_kernel<<<dim3(CN / 16, JN / 128), 256, 0, stream>>>(A16, Bt, lin_b, Value);

  tmp_max_kernel<<<2048, 256, 0, stream>>>(Value, rowv, degv, red);
  final_max_kernel<<<1, 1024, 0, stream>>>(red, maxv);
  tmp_expsum_kernel<<<2048, 256, 0, stream>>>(Value, rowv, degv, maxv, red);
  final_sum_kernel<<<1, 1024, 0, stream>>>(red, sumv);
  poss_kernel<<<4096, 256, 0, stream>>>(Value, rowv, degv, maxv, sumv, Poss);
}
